// MultiheadAttention_59313498357771
// MI455X (gfx1250) — matrix operand order repaired and checked by reading, not compiled, not run on hardware
//
#include <hip/hip_runtime.h>
#include <hip/hip_bf16.h>

// MI455X / gfx1250 multi-head attention:
//   proj_qkv  : f32 -> f16 tiled GEMM projections (q scaled by 1/8, V stored transposed)
//   attn_fa   : flash-attention streaming softmax, all-WMMA, async double-buffered K/V tiles
//   out_proj  : f16 GEMM vs Wo + bias -> f32 output
// Matrix math uses v_wmma_f32_16x16x32_f16 (wave32, f32 accumulation);
// K/V tile staging uses global_load_async_to_lds_b128 + s_wait_asynccnt.

typedef __attribute__((ext_vector_type(16))) _Float16 v16h;
typedef __attribute__((ext_vector_type(8)))  _Float16 h8;
typedef __attribute__((ext_vector_type(4)))  _Float16 h4;
typedef __attribute__((ext_vector_type(8)))  float    v8f;
typedef __attribute__((ext_vector_type(4)))  float    f4;
typedef __attribute__((ext_vector_type(4)))  int      v4i;

static constexpr int Bsz = 4, S = 2048, D = 1024, H = 16, DH = 64;

#ifndef __has_builtin
#define __has_builtin(x) 0
#endif
#if __has_builtin(__builtin_amdgcn_global_load_async_to_lds_b128) && \
    __has_builtin(__builtin_amdgcn_s_wait_asynccnt)
#define HAVE_ASYNC_LDS 1
#else
#define HAVE_ASYNC_LDS 0
#endif

// 16-byte global -> LDS copy. Async (ASYNCcnt-tracked) when the toolchain
// exposes the gfx1250 builtin; otherwise synchronous reg round-trip.
// ROCm clang-22 declares the builtin with int4* pointer parameters.
static __device__ inline void cp_g2s_16B(_Float16* lds, const _Float16* g) {
#if HAVE_ASYNC_LDS
  __builtin_amdgcn_global_load_async_to_lds_b128((v4i*)g, (v4i*)lds, 0, 0);
#else
  *(h8*)lds = *(const h8*)g;
#endif
}
static __device__ inline void cp_wait_all() {
#if HAVE_ASYNC_LDS
  __builtin_amdgcn_s_wait_asynccnt(0);
#endif
}

// --- WMMA helpers -----------------------------------------------------------

static __device__ inline v8f wmma_f16(v16h a, v16h b, v8f c) {
  // (neg_a, A, neg_b, B, c_mod, C, reuse_a, reuse_b)
  return __builtin_amdgcn_wmma_f32_16x16x32_f16(false, a, false, b, (short)0, c,
                                                false, false);
}

// A fragment (16x32, M x K): lane<16 holds K 0..15; lane>=16 holds K 16..31.
// Caller passes p = rowPtr + (lane>>4)*16 ; halves [0..7]=p[0..7], [8..15]=p[8..15].
static __device__ inline v16h load_frag_a(const _Float16* p) {
  h8 lo = *(const h8*)p;
  h8 hi = *(const h8*)(p + 8);
  return __builtin_shufflevector(lo, hi, 0, 1, 2, 3, 4, 5, 6, 7,
                                 8, 9, 10, 11, 12, 13, 14, 15);
}

// B fragment (32x16, K x N): lane<16 holds K 0..15, lane>=16 holds K 16..31,
// contiguous per lane. Caller passes p = colPtr + (lane>>4)*16.
static __device__ inline v16h load_frag_b(const _Float16* p) {
  h8 lo = *(const h8*)p;
  h8 hi = *(const h8*)(p + 8);
  return __builtin_shufflevector(lo, hi, 0, 1, 2, 3, 4, 5, 6, 7,
                                 8, 9, 10, 11, 12, 13, 14, 15);
}

// --- Kernel 1: QKV projections ---------------------------------------------
// grid = (64, 16, 3): x = 128-row block over B*S, y = head, z = {q,k,v}
// block = 256 (8 waves); wave computes a 16x64 output tile; K-steps of 32.

__global__ __launch_bounds__(256) void proj_qkv(
    const float* __restrict__ Xq, const float* __restrict__ Xk,
    const float* __restrict__ Xv, const float* __restrict__ Wq,
    const float* __restrict__ Wk, const float* __restrict__ Wv,
    const float* __restrict__ bq, const float* __restrict__ bk,
    const float* __restrict__ bv, _Float16* __restrict__ qh,
    _Float16* __restrict__ kh, _Float16* __restrict__ vT) {
  const int z = blockIdx.z;
  const int h = blockIdx.y;
  const int rowg0 = blockIdx.x * 128;

  const float* X = (z == 0) ? Xq : (z == 1) ? Xk : Xv;
  const float* W = ((z == 0) ? Wq : (z == 1) ? Wk : Wv) + (size_t)h * D * DH;
  const float* bias = ((z == 0) ? bq : (z == 1) ? bk : bv) + h * DH;

  __shared__ _Float16 As[128 * 40];  // 128 rows x 32 k, stride 40 (80B rows)
  __shared__ _Float16 Bs[64 * 40];   // Bs[n][k] (transposed weight tile)

  const int tid = threadIdx.x;
  const int w = tid >> 5;
  const int l = tid & 31;
  const int lh = l & 15;
  const int lg = l >> 4;

  v8f acc[4] = {{}, {}, {}, {}};

  for (int kt = 0; kt < 32; ++kt) {
    const int kb = kt * 32;
    __syncthreads();
    // A tile: 128x32 f32 -> f16 (float4 per slot, packed converts)
#pragma unroll
    for (int i = 0; i < 4; ++i) {
      int slot = tid + i * 256;  // 0..1023
      int r = slot >> 3;
      int c = (slot & 7) * 4;
      f4 xv = *(const f4*)(X + (size_t)(rowg0 + r) * D + kb + c);
      h4 hv;
      hv[0] = (_Float16)xv[0]; hv[1] = (_Float16)xv[1];
      hv[2] = (_Float16)xv[2]; hv[3] = (_Float16)xv[3];
      *(h4*)(As + r * 40 + c) = hv;
    }
    // B tile: 32x64 f32, float4 along n (coalesced b128 loads), then
    // transpose-scatter into Bs[n][k] with b16 LDS stores.
#pragma unroll
    for (int i = 0; i < 2; ++i) {
      int slot = tid + i * 256;  // 0..511
      int k = slot >> 4;         // 0..31
      int n4 = (slot & 15) * 4;  // 0..60
      f4 wv = *(const f4*)(W + (size_t)(kb + k) * DH + n4);
#pragma unroll
      for (int j = 0; j < 4; ++j)
        Bs[(n4 + j) * 40 + k] = (_Float16)wv[j];
    }
    // Prefetch next K-step's A tile (global_prefetch_b8) while we compute.
    if (kt + 1 < 32)
      __builtin_prefetch(X + (size_t)(rowg0 + (tid >> 3)) * D + kb + 32 +
                             (tid & 7) * 4,
                         0, 1);
    __syncthreads();

    v16h a = load_frag_a(As + (w * 16 + lh) * 40 + lg * 16);
#pragma unroll
    for (int nf = 0; nf < 4; ++nf) {
      v16h b = load_frag_b(Bs + (nf * 16 + lh) * 40 + lg * 16);
      acc[nf] = wmma_f16(a, b, acc[nf]);
    }
  }

  // Epilogue: bias (+ 1/sqrt(dk) folded into q), store f16.
  float bi[4];
#pragma unroll
  for (int nf = 0; nf < 4; ++nf) bi[nf] = bias[nf * 16 + lh];

#pragma unroll
  for (int r = 0; r < 8; ++r) {
    int m = w * 16 + lg * 8 + r;
    int rowg = rowg0 + m;
    int bb = rowg >> 11;       // / S
    int s = rowg & (S - 1);    // % S
#pragma unroll
    for (int nf = 0; nf < 4; ++nf) {
      int n = nf * 16 + lh;
      float v = acc[nf][r] + bi[nf];
      if (z == 0) {
        qh[((size_t)(bb * H + h) * S + s) * DH + n] = (_Float16)(v * 0.125f);
      } else if (z == 1) {
        kh[((size_t)(bb * H + h) * S + s) * DH + n] = (_Float16)v;
      } else {
        vT[((size_t)(bb * H + h) * DH + n) * S + s] = (_Float16)v;
      }
    }
  }
}

// --- Kernel 2: flash attention ---------------------------------------------
// grid = (16, 16, 4): x = 128-q-row block, y = head, z = batch. block = 256.
// Wave owns 16 q-rows; streams 32 key blocks of 64 with double-buffered,
// async-staged K / V^T LDS tiles; online softmax kept in registers.

__global__ __launch_bounds__(256) void attn_fa(
    const _Float16* __restrict__ qh, const _Float16* __restrict__ kh,
    const _Float16* __restrict__ vT, _Float16* __restrict__ attnOut) {
  const int b = blockIdx.z, h = blockIdx.y, qblk = blockIdx.x;
  const int tid = threadIdx.x;
  const int w = tid >> 5;
  const int l = tid & 31;
  const int lh = l & 15;
  const int lg = l >> 4;

  __shared__ _Float16 Ks[2][64 * 72];   // Ks[buf][key][feature]
  __shared__ _Float16 Vts[2][64 * 72];  // Vts[buf][feature][key]
  __shared__ _Float16 Ps[8 * 16 * 72];  // per-wave P tile (16 x 64)

  const size_t bh = (size_t)(b * H + h);
  const _Float16* qbase = qh + (bh * S + qblk * 128) * DH;
  const _Float16* kbase = kh + bh * S * DH;
  const _Float16* vbase = vT + bh * DH * S;

  // Stage one 64-key K tile + V^T tile into LDS buffer `buf` (async DMA).
  auto stage = [&](int buf, int jb) {
#pragma unroll
    for (int i = 0; i < 2; ++i) {
      int slot = tid + i * 256;  // 0..511
      int r8 = slot >> 3;        // row (key for K, feature for V^T)
      int c8 = (slot & 7) * 8;   // 8-half column chunk
      cp_g2s_16B(&Ks[buf][r8 * 72 + c8],
                 kbase + (size_t)(jb * 64 + r8) * DH + c8);
      cp_g2s_16B(&Vts[buf][r8 * 72 + c8],
                 vbase + (size_t)r8 * S + jb * 64 + c8);
    }
  };

  // Q fragments (held in registers for the whole loop)
  const _Float16* qrow = qbase + (size_t)(w * 16 + lh) * DH + lg * 16;
  v16h qa0 = load_frag_a(qrow);        // features 0..31
  v16h qa1 = load_frag_a(qrow + 32);   // features 32..63

  v8f oacc[4] = {{}, {}, {}, {}};
  float mr[8], lr[8];
#pragma unroll
  for (int r = 0; r < 8; ++r) { mr[r] = -3.0e38f; lr[r] = 0.f; }

  _Float16* Pw = Ps + w * 16 * 72;

  constexpr int NB = S / 64;
  stage(0, 0);

  for (int jb = 0; jb < NB; ++jb) {
    const int cur = jb & 1;
    cp_wait_all();       // this wave's async tile arrivals
    __syncthreads();     // everyone's tile is in LDS; prev buf fully consumed
    if (jb + 1 < NB) stage(cur ^ 1, jb + 1);  // overlap next tile with compute

    const _Float16* Kc = Ks[cur];
    const _Float16* Vc = Vts[cur];

    // Scores: 16 q-rows x 64 keys (4 key sub-blocks of 16)
    v8f sf[4] = {{}, {}, {}, {}};
#pragma unroll
    for (int nf = 0; nf < 4; ++nf) {
      v16h bk0 = load_frag_b(Kc + (nf * 16 + lh) * 72 + lg * 16);
      v16h bk1 = load_frag_b(Kc + (nf * 16 + lh) * 72 + 32 + lg * 16);
      sf[nf] = wmma_f16(qa0, bk0, sf[nf]);
      sf[nf] = wmma_f16(qa1, bk1, sf[nf]);
    }

    // Online softmax update (row reductions across 16-lane halves, wave32)
#pragma unroll
    for (int r = 0; r < 8; ++r) {
      float mx = fmaxf(fmaxf(sf[0][r], sf[1][r]), fmaxf(sf[2][r], sf[3][r]));
      mx = fmaxf(mx, __shfl_xor(mx, 1, 32));
      mx = fmaxf(mx, __shfl_xor(mx, 2, 32));
      mx = fmaxf(mx, __shfl_xor(mx, 4, 32));
      mx = fmaxf(mx, __shfl_xor(mx, 8, 32));
      float mnew = fmaxf(mr[r], mx);
      float alpha = __expf(mr[r] - mnew);
      mr[r] = mnew;
      float p0 = __expf(sf[0][r] - mnew);
      float p1 = __expf(sf[1][r] - mnew);
      float p2 = __expf(sf[2][r] - mnew);
      float p3 = __expf(sf[3][r] - mnew);
      float ls = p0 + p1 + p2 + p3;
      ls += __shfl_xor(ls, 1, 32);
      ls += __shfl_xor(ls, 2, 32);
      ls += __shfl_xor(ls, 4, 32);
      ls += __shfl_xor(ls, 8, 32);
      lr[r] = lr[r] * alpha + ls;
#pragma unroll
      for (int ff = 0; ff < 4; ++ff) oacc[ff][r] *= alpha;
      int m = lg * 8 + r;
      Pw[m * 72 + 0 * 16 + lh] = (_Float16)p0;
      Pw[m * 72 + 1 * 16 + lh] = (_Float16)p1;
      Pw[m * 72 + 2 * 16 + lh] = (_Float16)p2;
      Pw[m * 72 + 3 * 16 + lh] = (_Float16)p3;
    }

    // P x V (A from per-wave LDS tile; B from V^T tile)
    v16h pa0 = load_frag_a(Pw + lh * 72 + lg * 16);        // keys 0..31
    v16h pa1 = load_frag_a(Pw + lh * 72 + 32 + lg * 16);   // keys 32..63
#pragma unroll
    for (int ff = 0; ff < 4; ++ff) {
      v16h bv0 = load_frag_b(Vc + (ff * 16 + lh) * 72 + lg * 16);
      v16h bv1 = load_frag_b(Vc + (ff * 16 + lh) * 72 + 32 + lg * 16);
      oacc[ff] = wmma_f16(pa0, bv0, oacc[ff]);
      oacc[ff] = wmma_f16(pa1, bv1, oacc[ff]);
    }
    __syncthreads();  // all waves done reading buf[cur] before it is restaged
  }

  // Normalize and store in concat layout [B*S, H*DH] f16
#pragma unroll
  for (int r = 0; r < 8; ++r) {
    float inv = 1.0f / lr[r];
    int m = w * 16 + lg * 8 + r;
    size_t srow = (size_t)b * S + qblk * 128 + m;
    _Float16* dst = attnOut + srow * (H * DH) + h * DH;
#pragma unroll
    for (int ff = 0; ff < 4; ++ff)
      dst[ff * 16 + lh] = (_Float16)(oacc[ff][r] * inv);
  }
}

// --- Kernel 3: output projection -------------------------------------------
// grid = (64, 16): x = 128-row block over B*S, y = 64-col block over D.

__global__ __launch_bounds__(256) void out_proj(
    const _Float16* __restrict__ A, const float* __restrict__ Wo,
    const float* __restrict__ bo, float* __restrict__ out) {
  const int rowg0 = blockIdx.x * 128;
  const int nb0 = blockIdx.y * 64;

  __shared__ _Float16 As[128 * 40];
  __shared__ _Float16 Bs[64 * 40];

  const int tid = threadIdx.x;
  const int w = tid >> 5;
  const int l = tid & 31;
  const int lh = l & 15;
  const int lg = l >> 4;

  v8f acc[4] = {{}, {}, {}, {}};

  for (int kt = 0; kt < 32; ++kt) {
    const int kb = kt * 32;
    __syncthreads();
    // A tile (already f16): 128 x 32, async global->LDS copy
#pragma unroll
    for (int i = 0; i < 2; ++i) {
      int slot = tid + i * 256;  // 0..511
      int r = slot >> 2;
      int c8 = (slot & 3) * 8;
      cp_g2s_16B(As + r * 40 + c8,
                 A + (size_t)(rowg0 + r) * (H * DH) + kb + c8);
    }
    // B tile: Wo f32 32x64, float4 along n (coalesced b128 loads), then
    // transpose-scatter into Bs[n][k] f16.
#pragma unroll
    for (int i = 0; i < 2; ++i) {
      int slot = tid + i * 256;  // 0..511
      int k = slot >> 4;         // 0..31
      int n4 = (slot & 15) * 4;  // 0..60
      f4 wv = *(const f4*)(Wo + (size_t)(kb + k) * D + nb0 + n4);
#pragma unroll
      for (int j = 0; j < 4; ++j)
        Bs[(n4 + j) * 40 + k] = (_Float16)wv[j];
    }
    cp_wait_all();
    __syncthreads();

    v16h a = load_frag_a(As + (w * 16 + lh) * 40 + lg * 16);
#pragma unroll
    for (int nf = 0; nf < 4; ++nf) {
      v16h b = load_frag_b(Bs + (nf * 16 + lh) * 40 + lg * 16);
      acc[nf] = wmma_f16(a, b, acc[nf]);
    }
  }

  float bi[4];
#pragma unroll
  for (int nf = 0; nf < 4; ++nf) bi[nf] = bo[nb0 + nf * 16 + lh];

#pragma unroll
  for (int r = 0; r < 8; ++r) {
    int m = w * 16 + lg * 8 + r;
    size_t rowg = (size_t)rowg0 + m;
#pragma unroll
    for (int nf = 0; nf < 4; ++nf)
      out[rowg * D + nb0 + nf * 16 + lh] = acc[nf][r] + bi[nf];
  }
}

// --- Host launch ------------------------------------------------------------

extern "C" void kernel_launch(void* const* d_in, const int* in_sizes, int n_in,
                              void* d_out, int out_size, void* d_ws,
                              size_t ws_size, hipStream_t stream) {
  const float* queries = (const float*)d_in[0];
  const float* keys    = (const float*)d_in[1];
  const float* values  = (const float*)d_in[2];
  const float* Wq = (const float*)d_in[3];
  const float* bq = (const float*)d_in[4];
  const float* Wk = (const float*)d_in[5];
  const float* bk = (const float*)d_in[6];
  const float* Wv = (const float*)d_in[7];
  const float* bv = (const float*)d_in[8];
  const float* Wo = (const float*)d_in[9];
  const float* bo = (const float*)d_in[10];
  float* out = (float*)d_out;

  // Workspace layout (f16): qh | kh | vT | attn-concat, 16 MiB each = 64 MiB.
  const size_t elems = (size_t)Bsz * H * S * DH;  // 8,388,608
  char* ws = (char*)d_ws;
  _Float16* qh = (_Float16*)(ws);
  _Float16* kh = (_Float16*)(ws + elems * 2);
  _Float16* vT = (_Float16*)(ws + elems * 4);
  _Float16* at = (_Float16*)(ws + elems * 6);

  dim3 blk(256);
  proj_qkv<<<dim3(64, H, 3), blk, 0, stream>>>(queries, keys, values, Wq, Wk,
                                               Wv, bq, bk, bv, qh, kh, vT);
  attn_fa<<<dim3(S / 128, H, Bsz), blk, 0, stream>>>(qh, kh, vT, at);
  out_proj<<<dim3(64, D / 64), blk, 0, stream>>>(at, Wo, bo, out);
}
